// SpatialProcessor_45088566673698
// MI455X (gfx1250) — compile-verified, matrix operand order checked by reading, not run on hardware
//
#include <hip/hip_runtime.h>
#include <hip/hip_bf16.h>

// ---------------------------------------------------------------------------
// GATv2 x2 layers on fully-connected graph. B=4, N=2048, H=4, D=32, Fin=32.
// Pipeline per layer:
//   P: xl = act@Wl + bl   (f16 out, layout [b*H+h][j][d])     -- WMMA f16
//      xr = act@Wr + br   (f32 out, layout [b*H+h][i][d])     -- WMMA f16
//      (W staged in LDS once per block, shared by 8 waves)
//   A: per (b,h, 128-target tile): stage full xl[b,h] (128KB f16) in LDS
//      via GLOBAL_LOAD_ASYNC_TO_LDS_B128 (ASYNCcnt), flash-style: per
//      32-source chunk compute scores in f32 VALU, exp (shift-invariant),
//      transpose probs via LDS, aggregate with 2x v_wmma_f32_16x16x32_f16
//      into f32 accumulators, normalize at end.
//   E: bias + LayerNorm (+ReLU layer1).
// ---------------------------------------------------------------------------

typedef _Float16 v16h __attribute__((ext_vector_type(16)));
typedef float    v8f  __attribute__((ext_vector_type(8)));
typedef int      v4i_gcc __attribute__((vector_size(16)));   // matches builtin sig

#define NNODES 2048
#define DHEAD  32
#define NCOLS  128   // H*D
#define EXP_SHIFT 4.0f

__device__ __forceinline__ v8f wmma_f16f32(v16h a, v16h b, v8f c) {
  return __builtin_amdgcn_wmma_f32_16x16x32_f16(
      /*neg_a=*/false, a, /*neg_b=*/false, b,
      /*c_mod=*/(short)0, c, /*reuse_a=*/false, /*reuse_b=*/false);
}

union HFrag {            // pun two 16B LDS loads into a v16h fragment
  v16h   v;
  float4 f4[2];
};

// ---------------------------------------------------------------------------
// Projection: out[(b*H+h)][row%N][d] = act[row]@W[:,h*32+d] + bias
// grid.x = (M/16 * 8) / 8 = 512 blocks, block = 256 (8 waves, 1 tile/wave)
// dynamic LDS: W staged (K*128 f32)
// ---------------------------------------------------------------------------
template <typename OutT>
__global__ __launch_bounds__(256) void gat_proj_kernel(
    const float* __restrict__ act,   // [M][K]
    const float* __restrict__ W,     // [K][128]
    const float* __restrict__ bias,  // [128]
    int K, OutT* __restrict__ outp)
{
  extern __shared__ char psm[];
  float* w_lds = (float*)psm;        // K*128 f32

  const int tid  = threadIdx.x;
  const int lane = tid & 31;
  const int wv   = tid >> 5;

  // stage W cooperatively (16KB for K=32, 64KB for K=128)
#pragma unroll 1
  for (int it = tid; it < K * 32; it += 256)   // K*128/4 float4s
    ((float4*)w_lds)[it] = ((const float4*)W)[it];
  __syncthreads();

  const int tile = blockIdx.x * 8 + wv;      // 4096 tiles total
  const int mt   = tile >> 3;                // 512 row tiles
  const int nt   = tile & 7;                 // 8 col tiles
  const int kb   = (lane < 16) ? 0 : 8;
  const int mrow = lane & 15;
  const int n    = nt * 16 + (lane & 15);

  v8f c = {};
  for (int ks = 0; ks < K; ks += 32) {
    const float* ap = act + (size_t)(mt * 16 + mrow) * K + ks + kb;
    __builtin_prefetch(ap + K * 16, 0, 0);
    float4 a0 = *(const float4*)(ap);
    float4 a1 = *(const float4*)(ap + 4);
    float4 a2 = *(const float4*)(ap + 16);
    float4 a3 = *(const float4*)(ap + 20);
    v16h A;
    A[0]=(_Float16)a0.x; A[1]=(_Float16)a0.y; A[2]=(_Float16)a0.z; A[3]=(_Float16)a0.w;
    A[4]=(_Float16)a1.x; A[5]=(_Float16)a1.y; A[6]=(_Float16)a1.z; A[7]=(_Float16)a1.w;
    A[8]=(_Float16)a2.x; A[9]=(_Float16)a2.y; A[10]=(_Float16)a2.z; A[11]=(_Float16)a2.w;
    A[12]=(_Float16)a3.x; A[13]=(_Float16)a3.y; A[14]=(_Float16)a3.z; A[15]=(_Float16)a3.w;
    v16h Bf;
#pragma unroll
    for (int t = 0; t < 8; ++t) {
      Bf[t]     = (_Float16)w_lds[(ks + kb + t) * NCOLS + n];
      Bf[t + 8] = (_Float16)w_lds[(ks + kb + 16 + t) * NCOLS + n];
    }
    c = wmma_f16f32(A, Bf, c);
  }

  const float bv   = bias[n];
  const int   mb   = (lane < 16) ? 0 : 8;
  const int   hcol = n >> 5;
  const int   d    = n & 31;
#pragma unroll
  for (int r = 0; r < 8; ++r) {
    const int row  = mt * 16 + mb + r;
    const int bidx = row >> 11;         // row / 2048
    const int jj   = row & (NNODES - 1);
    const size_t o = (((size_t)bidx * 4 + hcol) * NNODES + jj) * DHEAD + d;
    outp[o] = (OutT)(c[r] + bv);
  }
}

// ---------------------------------------------------------------------------
// Attention: grid = (16 target tiles of 128, B*H). block=256 (8 waves x 16 i)
// dynamic LDS: xl (128KB f16) + xr tiles (16KB) + w tiles (8KB) + att (128B)
// LDS caps occupancy at 2 blocks/WGP -> allow big VGPR budget (no spills).
// ---------------------------------------------------------------------------
__global__ __launch_bounds__(256, 2) void gat_attn_kernel(
    const _Float16* __restrict__ xl,  // [BH][N][32] f16
    const float*    __restrict__ xr,  // [BH][N][32] f32
    const float*    __restrict__ att, // [H][32]
    float*          __restrict__ out) // [B][N][128]
{
  extern __shared__ char smem[];
  _Float16* xl_sh  = (_Float16*)smem;                        // 2048*32 halves
  float*    xr_sh  = (float*)(smem + 131072);                // 8*512 f32
  _Float16* w_sh   = (_Float16*)(smem + 131072 + 16384);     // 8*512 halves
  float*    att_sh = (float*)(smem + 131072 + 16384 + 8192); // 32 f32

  const int tid  = threadIdx.x;
  const int lane = tid & 31;
  const int wv   = tid >> 5;
  const int bh   = blockIdx.y;
  const int b    = bh >> 2;
  const int h    = bh & 3;

  // stage full xl[b,h] into LDS (128KB): async DMA path when available
  {
    const _Float16* srcp = xl + (size_t)bh * NNODES * DHEAD;
#if __has_builtin(__builtin_amdgcn_global_load_async_to_lds_b128)
    typedef v4i_gcc __attribute__((address_space(1))) g_v4i;
    typedef v4i_gcc __attribute__((address_space(3))) l_v4i;
    g_v4i* gsrc = (g_v4i*)srcp;
    l_v4i* ldst = (l_v4i*)xl_sh;
#pragma unroll 1
    for (int it = tid; it < NNODES * DHEAD / 8; it += 256)
      __builtin_amdgcn_global_load_async_to_lds_b128(gsrc + it, ldst + it, 0, 0);
#if __has_builtin(__builtin_amdgcn_s_wait_asynccnt)
    __builtin_amdgcn_s_wait_asynccnt(0);
#else
    asm volatile("s_wait_asynccnt 0" ::: "memory");
#endif
#else
    const float4* src4 = (const float4*)srcp;
    float4* dst4 = (float4*)xl_sh;
#pragma unroll 1
    for (int it = tid; it < NNODES * DHEAD / 8; it += 256) {
      __builtin_prefetch(src4 + it + 512, 0, 0);
      dst4[it] = src4[it];
    }
#endif
  }
  if (tid < 32) att_sh[tid] = att[h * DHEAD + tid];

  const int i0 = blockIdx.x * 128 + wv * 16;   // this wave's 16 targets
  {
    const float* xrg = xr + ((size_t)bh * NNODES + i0) * DHEAD;
#pragma unroll
    for (int t = 0; t < 16; ++t)
      xr_sh[wv * 512 + t * 32 + lane] = xrg[t * 32 + lane];
  }
  __syncthreads();

  float s[16];
#pragma unroll
  for (int i = 0; i < 16; ++i) s[i] = 0.0f;
  v8f c0 = {};
  v8f c1 = {};
  const int kb  = (lane < 16) ? 0 : 8;
  const int nlo = lane & 15;

#pragma unroll 1
  for (int jb = 0; jb < NNODES; jb += 32) {
    const int j = jb + lane;

    float e[16];
#pragma unroll
    for (int i = 0; i < 16; ++i) e[i] = 0.0f;
    // e[i] = sum_d att[d] * leaky_relu(xl[j][d] + xr[i][d])
    //      = sum_d (0.6*att)*z + (0.4*att)*|z|
    // runtime dg-loop (4 groups of 8 d's): keeps att/x values from being
    // hoisted into ~64 live VGPRs across the jb loop (spill-free).
#pragma unroll 1
    for (int dg = 0; dg < 4; ++dg) {
      const float4 xv4 = *(const float4*)&xl_sh[j * DHEAD + dg * 8]; // 8 halves
      const _Float16* xh = (const _Float16*)&xv4;
#pragma unroll
      for (int dd = 0; dd < 8; ++dd) {
        const int   d  = dg * 8 + dd;
        const float xv = (float)xh[dd];
        const float ad = att_sh[d];
        const float ca = 0.6f * ad;
        const float cb = 0.4f * ad;
#pragma unroll
        for (int i = 0; i < 16; ++i) {
          const float z = xv + xr_sh[wv * 512 + i * 32 + d];
          e[i] += ca * z + cb * fabsf(z);
        }
      }
    }
    // shift-invariant softmax numerators; stash f16 probs for WMMA A-frag
#pragma unroll
    for (int i = 0; i < 16; ++i) {
      const float w = __expf(e[i] - EXP_SHIFT);
      s[i] += w;
      w_sh[wv * 512 + i * 32 + lane] = (_Float16)w;
    }
    __builtin_amdgcn_wave_barrier();

    // A-frag: two contiguous 16B runs per lane (row nlo, K = kb+{0..7,16..23})
    HFrag af;
    af.f4[0] = *(const float4*)&w_sh[wv * 512 + nlo * 32 + kb];
    af.f4[1] = *(const float4*)&w_sh[wv * 512 + nlo * 32 + kb + 16];

    v16h B0, B1;
#pragma unroll
    for (int t = 0; t < 8; ++t) {
      const int j1 = jb + kb + t;
      const int j2 = jb + kb + 16 + t;
      B0[t]     = xl_sh[j1 * DHEAD + nlo];
      B0[t + 8] = xl_sh[j2 * DHEAD + nlo];
      B1[t]     = xl_sh[j1 * DHEAD + 16 + nlo];
      B1[t + 8] = xl_sh[j2 * DHEAD + 16 + nlo];
    }
    c0 = wmma_f16f32(af.v, B0, c0);   // out cols d = 0..15
    c1 = wmma_f16f32(af.v, B1, c1);   // out cols d = 16..31
    __builtin_amdgcn_wave_barrier();
  }

  // full denominators: reduce partials across all 32 lanes
#pragma unroll
  for (int off = 16; off > 0; off >>= 1) {
#pragma unroll
    for (int i = 0; i < 16; ++i) s[i] += __shfl_xor(s[i], off, 32);
  }

#pragma unroll
  for (int r = 0; r < 8; ++r) {
    const int   i   = i0 + ((lane < 16) ? 0 : 8) + r;
    const float inv = 1.0f / ((lane < 16) ? s[r] : s[r + 8]);
    float* o = out + ((size_t)b * NNODES + i) * NCOLS + h * DHEAD;
    o[nlo]      = c0[r] * inv;
    o[16 + nlo] = c1[r] * inv;
  }
}

// ---------------------------------------------------------------------------
// Epilogue: out = LN(in + bias) * g + b, optional ReLU. One row per wave.
// ---------------------------------------------------------------------------
__global__ __launch_bounds__(256) void gat_ln_kernel(
    const float* __restrict__ in, const float* __restrict__ bias,
    const float* __restrict__ g,  const float* __restrict__ bb,
    float* __restrict__ out, int do_relu)
{
  const int lane = threadIdx.x & 31;
  const int wv   = threadIdx.x >> 5;
  const int row  = blockIdx.x * 8 + wv;   // 8192 rows

  float4 v = *(const float4*)(in + (size_t)row * NCOLS + lane * 4);
  const float4 bi = *(const float4*)(bias + lane * 4);
  v.x += bi.x; v.y += bi.y; v.z += bi.z; v.w += bi.w;

  float sum = v.x + v.y + v.z + v.w;
#pragma unroll
  for (int off = 16; off > 0; off >>= 1) sum += __shfl_xor(sum, off, 32);
  const float mu = sum * (1.0f / 128.0f);

  const float d0 = v.x - mu, d1 = v.y - mu, d2 = v.z - mu, d3 = v.w - mu;
  float q = d0 * d0 + d1 * d1 + d2 * d2 + d3 * d3;
#pragma unroll
  for (int off = 16; off > 0; off >>= 1) q += __shfl_xor(q, off, 32);
  const float rs = rsqrtf(q * (1.0f / 128.0f) + 1e-5f);

  const float4 gg = *(const float4*)(g + lane * 4);
  const float4 b2 = *(const float4*)(bb + lane * 4);
  float4 y;
  y.x = d0 * rs * gg.x + b2.x;
  y.y = d1 * rs * gg.y + b2.y;
  y.z = d2 * rs * gg.z + b2.z;
  y.w = d3 * rs * gg.w + b2.w;
  if (do_relu) {
    y.x = fmaxf(y.x, 0.0f); y.y = fmaxf(y.y, 0.0f);
    y.z = fmaxf(y.z, 0.0f); y.w = fmaxf(y.w, 0.0f);
  }
  *(float4*)(out + (size_t)row * NCOLS + lane * 4) = y;
}

// ---------------------------------------------------------------------------
extern "C" void kernel_launch(void* const* d_in, const int* in_sizes, int n_in,
                              void* d_out, int out_size, void* d_ws, size_t ws_size,
                              hipStream_t stream) {
  (void)in_sizes; (void)n_in; (void)out_size; (void)ws_size;
  const float* x     = (const float*)d_in[0];
  // d_in[1] = embedding: only defines (dense) connectivity -> unused
  const float* Wl1   = (const float*)d_in[2];
  const float* bl1   = (const float*)d_in[3];
  const float* Wr1   = (const float*)d_in[4];
  const float* br1   = (const float*)d_in[5];
  const float* att1  = (const float*)d_in[6];
  const float* bias1 = (const float*)d_in[7];
  const float* ln1g  = (const float*)d_in[8];
  const float* ln1b  = (const float*)d_in[9];
  const float* Wl2   = (const float*)d_in[10];
  const float* bl2   = (const float*)d_in[11];
  const float* Wr2   = (const float*)d_in[12];
  const float* br2   = (const float*)d_in[13];
  const float* att2  = (const float*)d_in[14];
  const float* bias2 = (const float*)d_in[15];
  const float* ln2g  = (const float*)d_in[16];
  const float* ln2b  = (const float*)d_in[17];

  char* ws = (char*)d_ws;
  _Float16* xlh  = (_Float16*)ws;                       // 8192*128 f16 = 2MB
  float*    xrb  = (float*)(ws + (2ull << 20));         // 8192*128 f32 = 4MB
  float*    attb = (float*)(ws + (6ull << 20));         // 4MB
  float*    hbuf = (float*)(ws + (10ull << 20));        // 4MB

  const dim3 blk(256);
  const dim3 gP(512);                 // 4096 tiles / 8 waves
  const dim3 gA(16, 16);              // 16 target tiles x (B*H)
  const dim3 gE(1024);                // 8192 rows / 8 waves
  const size_t smemA  = 131072 + 16384 + 8192 + 128;    // 152.1 KB < 320 KB
  const size_t smemP1 = 32 * NCOLS * sizeof(float);     // 16 KB
  const size_t smemP2 = 128 * NCOLS * sizeof(float);    // 64 KB

  // ----- layer 1 -----
  gat_proj_kernel<_Float16><<<gP, blk, smemP1, stream>>>(x, Wl1, bl1, 32, xlh);
  gat_proj_kernel<float>   <<<gP, blk, smemP1, stream>>>(x, Wr1, br1, 32, xrb);
  gat_attn_kernel<<<gA, blk, smemA, stream>>>(xlh, xrb, att1, attb);
  gat_ln_kernel  <<<gE, blk, 0, stream>>>(attb, bias1, ln1g, ln1b, hbuf, 1);

  // ----- layer 2 -----
  gat_proj_kernel<_Float16><<<gP, blk, smemP2, stream>>>(hbuf, Wl2, bl2, 128, xlh);
  gat_proj_kernel<float>   <<<gP, blk, smemP2, stream>>>(hbuf, Wr2, br2, 128, xrb);
  gat_attn_kernel<<<gA, blk, smemA, stream>>>(xlh, xrb, att2, attb);
  gat_ln_kernel  <<<gE, blk, 0, stream>>>(attb, bias2, ln2g, ln2b, (float*)d_out, 0);
}
